// ClippedGRU_3272765080208
// MI455X (gfx1250) — matrix operand order repaired and checked by reading, not compiled, not run on hardware
//
#include <hip/hip_runtime.h>
#include <cstdint>

// ---------------------------------------------------------------------------
// ClippedGRU for MI455X (gfx1250, wave32, WMMA bf16 16x16x32, f32 accumulate)
//
// Shapes: B=256, T=400, I=256, H=512, G=3H=1536.
// Pipeline:
//   K1 prep : f32 -> bf16 conversion of x, W_ih, W_hh (into workspace)
//   K2 xgemm: x_all[t][b][g] = x @ W_ih^T + b_ih   (WMMA bf16, f32 out)
//   K3 scan : persistent per-batch-row-group GRU recurrence, h in LDS,
//             no inter-workgroup sync needed (batch rows are independent).
//             x_all for step t+1 is software-prefetched (global_prefetch_b8)
//             during step t's WMMA work to hide HBM latency on the
//             latency-bound sequential scan.
// ---------------------------------------------------------------------------

#define B_  256
#define T_  400
#define I_  256
#define H_  512
#define G_  1536
#define HPAD 520   // LDS row stride in bf16 elems (512 + 8 pad -> kills bank conflicts)

typedef __attribute__((ext_vector_type(16))) __bf16 v16bf;
typedef __attribute__((ext_vector_type(8)))  float  v8f;
typedef __attribute__((ext_vector_type(4)))  int    v4i;

union FragBF { v16bf v; v4i q[2]; };

__device__ __forceinline__ unsigned short f2bf(float f) {
    unsigned int u = __float_as_uint(f);
    unsigned int r = u + 0x7FFFu + ((u >> 16) & 1u);   // round-to-nearest-even
    return (unsigned short)(r >> 16);
}
__device__ __forceinline__ float bf2f(unsigned short h) {
    return __uint_as_float(((unsigned int)h) << 16);
}
__device__ __forceinline__ float sigmf(float x) { return 1.0f / (1.0f + __expf(-x)); }

__device__ __forceinline__ v8f wmma_bf16(v16bf a, v16bf b, v8f c) {
    // D = A(16x32 bf16) * B(32x16 bf16) + C(16x16 f32)
    return __builtin_amdgcn_wmma_f32_16x16x32_bf16(
        /*neg_a=*/false, a, /*neg_b=*/false, b,
        /*c_mod=*/(short)0, c, /*reuse_a=*/false, /*reuse_b=*/false);
}

// A-fragment (16x32 bf16) from a row-major bf16 matrix, 16B-aligned chunks.
// ISA layout: lane<16 -> K{0..7} then K{16..23}; lane>=16 -> K{8..15}, K{24..31}.
__device__ __forceinline__ v16bf load_a_rowmajor(const unsigned short* row_base,
                                                 int k, int lane) {
    FragBF f;
    const unsigned short* p = row_base + k + ((lane < 16) ? 0 : 8);
    f.q[0] = *(const v4i*)(p);
    f.q[1] = *(const v4i*)(p + 16);
    return f.v;
}

// B-fragment (32x16 bf16): lane holds column n = (lane&15) of W[n][k..],
// lane<16 -> K{0..7} then K{16..23}; lane>=16 -> K{8..15}, K{24..31}.
// W stored as the ORIGINAL [g][h] row-major layout -> contiguous loads.
__device__ __forceinline__ v16bf load_b_w(const unsigned short* W, int ldw,
                                          int nrow, int k, int lane) {
    FragBF f;
    const unsigned short* p = W + (size_t)nrow * ldw + k + ((lane < 16) ? 0 : 8);
    f.q[0] = *(const v4i*)(p);
    f.q[1] = *(const v4i*)(p + 16);
    return f.v;
}

// ---------------------------------------------------------------------------
// K1: f32 -> bf16 conversions (grid-stride)
// ---------------------------------------------------------------------------
__global__ void gru_prep(const float* __restrict__ x,
                         const float* __restrict__ Wih,
                         const float* __restrict__ Whh,
                         unsigned short* __restrict__ xbf,
                         unsigned short* __restrict__ Wihb,
                         unsigned short* __restrict__ Whhb) {
    size_t i = (size_t)blockIdx.x * blockDim.x + threadIdx.x;
    size_t stride = (size_t)gridDim.x * blockDim.x;
    for (size_t j = i; j < (size_t)B_ * T_ * I_; j += stride) xbf[j]  = f2bf(x[j]);
    for (size_t j = i; j < (size_t)G_ * I_;      j += stride) Wihb[j] = f2bf(Wih[j]);
    for (size_t j = i; j < (size_t)G_ * H_;      j += stride) Whhb[j] = f2bf(Whh[j]);
}

// ---------------------------------------------------------------------------
// K2: x_all[t][b][g] = x[b][t][:] @ W_ih[g][:] + b_ih[g]
// grid = (48 tiles, 400 t); block = 256 (8 waves). WG tile 64x128,
// wave tile 32x32 (2x2 WMMA tiles), K = 256 (8 k-steps of 32).
// ---------------------------------------------------------------------------
__global__ __launch_bounds__(256) void gru_xgemm(
    const unsigned short* __restrict__ xbf,
    const unsigned short* __restrict__ Wihb,
    const float* __restrict__ bias_ih,
    float* __restrict__ xall) {

    const int t    = blockIdx.y;
    const int tb   = blockIdx.x / 12;          // 4 row groups of 64
    const int tn   = blockIdx.x % 12;          // 12 col groups of 128
    const int lane = threadIdx.x & 31;
    const int wv   = threadIdx.x >> 5;
    const int wm   = wv >> 2;                  // 0..1
    const int wn   = wv & 3;                   // 0..3
    const int Rb   = tb * 64 + wm * 32;        // wave row base (batch dim)
    const int Cb   = tn * 128 + wn * 32;       // wave col base (gate dim)
    const int cn   = lane & 15;
    const int half8 = (lane < 16) ? 0 : 8;

    v8f acc[4];                                 // [mt*2+nt]
    const v8f vzero = {0.f,0.f,0.f,0.f,0.f,0.f,0.f,0.f};
#pragma unroll
    for (int i = 0; i < 4; ++i) acc[i] = vzero;

#pragma unroll 1
    for (int k = 0; k < I_; k += 32) {
        v16bf a0 = load_a_rowmajor(xbf + ((size_t)(Rb + cn)      * T_ + t) * I_, k, lane);
        v16bf a1 = load_a_rowmajor(xbf + ((size_t)(Rb + 16 + cn) * T_ + t) * I_, k, lane);
        v16bf b0 = load_b_w(Wihb, I_, Cb + cn,      k, lane);
        v16bf b1 = load_b_w(Wihb, I_, Cb + 16 + cn, k, lane);
        acc[0] = wmma_bf16(a0, b0, acc[0]);
        acc[1] = wmma_bf16(a0, b1, acc[1]);
        acc[2] = wmma_bf16(a1, b0, acc[2]);
        acc[3] = wmma_bf16(a1, b1, acc[3]);
    }

#pragma unroll
    for (int mt = 0; mt < 2; ++mt) {
#pragma unroll
        for (int nt = 0; nt < 2; ++nt) {
            int g = Cb + nt * 16 + cn;
            float bi = bias_ih[g];
#pragma unroll
            for (int r = 0; r < 8; ++r) {
                int row = Rb + mt * 16 + r + half8;         // batch index
                xall[((size_t)t * B_ + row) * G_ + g] = acc[mt * 2 + nt][r] + bi;
            }
        }
    }
}

// ---------------------------------------------------------------------------
// K3: persistent recurrence. 16 WGs; WG owns 16 batch rows for all 400 steps.
// h (16x512 bf16) lives in LDS; each of 8 waves owns 64 hidden columns so
// gate columns z_j / r_j / n_j for the same j stay in one wave -> only
// __syncthreads() between the two per-step matmuls. No global sync at all.
// ---------------------------------------------------------------------------

// Prefetch one step's x_all working set for this wave (3 gates x 16 rows x
// 4 cachelines of 64B = 192 lines = 6 per lane). Lowers to global_prefetch_b8.
__device__ __forceinline__ void prefetch_xall_step(const float* __restrict__ xall,
                                                   int t, int Rb, int colw, int lane) {
    const float* base = xall + ((size_t)t * B_ + Rb) * G_ + colw;
#pragma unroll
    for (int i = 0; i < 6; ++i) {
        int p    = lane + 32 * i;       // 0..191
        int gate = p >> 6;              // 0..2
        int m    = (p >> 2) & 15;       // 0..15
        int line = p & 3;               // 0..3
        __builtin_prefetch(base + (size_t)m * G_ + gate * H_ + line * 16, 0, 1);
    }
}

__global__ __launch_bounds__(256) void gru_scan(
    const float* __restrict__ xall,
    const unsigned short* __restrict__ Whhb,   // [1536][512] bf16 (z|r|n rows)
    const float* __restrict__ h0,
    const float* __restrict__ bias_hh,
    float* __restrict__ out,                   // [B][T][H] f32
    float* __restrict__ hlast) {               // [B][H] f32

    __shared__ unsigned short h_l [16][HPAD];
    __shared__ unsigned short rh_l[16][HPAD];

    const int tid   = threadIdx.x;
    const int lane  = tid & 31;
    const int wv    = tid >> 5;
    const int Rb    = blockIdx.x * 16;          // owned batch rows
    const int cn    = lane & 15;
    const int half8 = (lane < 16) ? 0 : 8;
    const int colw  = wv * 64;                  // wave's hidden-column base
    const v8f vzero = {0.f,0.f,0.f,0.f,0.f,0.f,0.f,0.f};

    // warm the cache for step 0's activations while h0 is staged into LDS
    prefetch_xall_step(xall, 0, Rb, colw, lane);

    for (int i = tid; i < 16 * H_; i += 256) {
        int m = i >> 9, j = i & 511;
        h_l[m][j] = f2bf(h0[(size_t)(Rb + m) * H_ + j]);
    }
    __syncthreads();

    float bz[4], br[4], bn[4];
#pragma unroll
    for (int nt = 0; nt < 4; ++nt) {
        int col = colw + nt * 16 + cn;
        bz[nt] = bias_hh[col];
        br[nt] = bias_hh[H_ + col];
        bn[nt] = bias_hh[2 * H_ + col];
    }

#pragma unroll 1
    for (int t = 0; t < T_; ++t) {
        // prefetch NEXT step's activations; the two WMMA k-loops below give
        // ~a full step of lead time to cover HBM latency.
        if (t + 1 < T_) prefetch_xall_step(xall, t + 1, Rb, colw, lane);

        // ---- phase 1: zr = h @ W_zr^T ----
        v8f az[4], ar[4];
#pragma unroll
        for (int nt = 0; nt < 4; ++nt) { az[nt] = vzero; ar[nt] = vzero; }

#pragma unroll 1
        for (int k = 0; k < H_; k += 32) {
            v16bf a = load_a_rowmajor(&h_l[cn][0], k, lane);
#pragma unroll
            for (int nt = 0; nt < 4; ++nt) {
                v16bf b = load_b_w(Whhb, H_, colw + nt * 16 + cn, k, lane);
                az[nt] = wmma_bf16(a, b, az[nt]);
            }
#pragma unroll
            for (int nt = 0; nt < 4; ++nt) {
                v16bf b = load_b_w(Whhb, H_, H_ + colw + nt * 16 + cn, k, lane);
                ar[nt] = wmma_bf16(a, b, ar[nt]);
            }
        }

        // ---- phase 1.5: z, r, rh = r*h (own columns only) ----
#pragma unroll
        for (int nt = 0; nt < 4; ++nt) {
            int col = colw + nt * 16 + cn;
#pragma unroll
            for (int r = 0; r < 8; ++r) {
                int m = r + half8;
                size_t xrow = ((size_t)t * B_ + (Rb + m)) * G_;
                float z  = sigmf(xall[xrow + col]        + az[nt][r] + bz[nt]);
                float rr = sigmf(xall[xrow + H_ + col]   + ar[nt][r] + br[nt]);
                float hv = bf2f(h_l[m][col]);
                rh_l[m][col] = f2bf(rr * hv);
                az[nt][r] = z;                    // keep z for phase 2
            }
        }
        __syncthreads();                          // rh complete for all waves

        // ---- phase 2: n_pre = (r*h) @ W_n^T ----
        v8f an[4];
#pragma unroll
        for (int nt = 0; nt < 4; ++nt) an[nt] = vzero;

#pragma unroll 1
        for (int k = 0; k < H_; k += 32) {
            v16bf a = load_a_rowmajor(&rh_l[cn][0], k, lane);
#pragma unroll
            for (int nt = 0; nt < 4; ++nt) {
                v16bf b = load_b_w(Whhb, H_, 2 * H_ + colw + nt * 16 + cn, k, lane);
                an[nt] = wmma_bf16(a, b, an[nt]);
            }
        }

        // ---- phase 2.5: n, h_new, clip, write out + update LDS h ----
#pragma unroll
        for (int nt = 0; nt < 4; ++nt) {
            int col = colw + nt * 16 + cn;
#pragma unroll
            for (int r = 0; r < 8; ++r) {
                int m = r + half8;
                int b = Rb + m;
                size_t xrow = ((size_t)t * B_ + b) * G_;
                float n  = tanhf(xall[xrow + 2 * H_ + col] + an[nt][r] + bn[nt]);
                float hv = bf2f(h_l[m][col]);
                float z  = az[nt][r];
                float hn = z * hv + (1.0f - z) * n;
                hn = fminf(5.0f, fmaxf(-5.0f, hn));
                out[((size_t)b * T_ + t) * H_ + col] = hn;
                h_l[m][col] = f2bf(hn);
            }
        }
        __syncthreads();                          // h complete before next step
    }

    for (int i = tid; i < 16 * H_; i += 256) {
        int m = i >> 9, j = i & 511;
        hlast[(size_t)(Rb + m) * H_ + j] = bf2f(h_l[m][j]);
    }
}

// ---------------------------------------------------------------------------
// Host-side launch. Workspace layout (bytes):
//   [0)           W_ih bf16   : 1536*256*2  =    786,432
//   [786432)      W_hh bf16   : 1536*512*2  =  1,572,864
//   [2359296)     x    bf16   : 256*400*256*2 = 52,428,800
//   [54788096)    x_all f32   : 400*256*1536*4 = 629,145,600   (t-major)
// total ~ 684 MB (assumes ws_size is sized generously).
// ---------------------------------------------------------------------------
extern "C" void kernel_launch(void* const* d_in, const int* in_sizes, int n_in,
                              void* d_out, int out_size, void* d_ws, size_t ws_size,
                              hipStream_t stream) {
    const float* x       = (const float*)d_in[0];
    const float* h0      = (const float*)d_in[1];
    const float* Wih     = (const float*)d_in[2];
    const float* bias_ih = (const float*)d_in[3];
    const float* Whh     = (const float*)d_in[4];
    const float* bias_hh = (const float*)d_in[5];

    char* ws = (char*)d_ws;
    unsigned short* Wihb = (unsigned short*)(ws + 0);
    unsigned short* Whhb = (unsigned short*)(ws + 786432);
    unsigned short* xbf  = (unsigned short*)(ws + 2359296);
    float*          xall = (float*)        (ws + 54788096);

    float* out   = (float*)d_out;
    float* hlast = out + (size_t)B_ * T_ * H_;

    gru_prep<<<2048, 256, 0, stream>>>(x, Wih, Whh, xbf, Wihb, Whhb);
    gru_xgemm<<<dim3(48, T_), 256, 0, stream>>>(xbf, Wihb, bias_ih, xall);
    gru_scan<<<16, 256, 0, stream>>>(xall, Whhb, h0, bias_hh, out, hlast);
}
